// MixtureOfSoftMaxACF_74466142978594
// MI455X (gfx1250) — matrix operand order repaired and checked by reading, not compiled, not run on hardware
//
#include <hip/hip_runtime.h>
#include <cstdint>
#include <cstddef>

// ---------------------------------------------------------------------------
// MixtureOfSoftMaxACF fused flash-attention for MI455X (gfx1250, wave32, WMMA)
// BS=8, N=N2=2048, D_K=128, M=2 (d=64), temperature = sqrt(128)
// - K/V pre-converted to bf16 in d_ws (one pass), then streamed into LDS via
//   CDNA5 async loads (global_load_async_to_lds_b128 + s_wait_asynccnt) with
//   double buffering.
// - Scores computed transposed (S^T = K_tile x Q^T) so softmax rows are lanes
//   and exp(P) lands directly in WMMA A-layout registers (no LDS round-trip).
// - O-accumulator rescale skipped (wave-uniform branch) when running max
//   did not grow.
// ---------------------------------------------------------------------------

typedef __attribute__((ext_vector_type(16))) __bf16 v16bf;
typedef __attribute__((ext_vector_type(8)))  float  v8f;

#define QSCALE 0.08838834764831845f /* 1/sqrt(128) */

union BF16x16 { unsigned short s[16]; unsigned u[8]; uint4 q[2]; v16bf v; };

__device__ __forceinline__ unsigned short bf1(float x) {
  union { __bf16 h; unsigned short s; } r;
  r.h = (__bf16)x;
  return r.s;
}
__device__ __forceinline__ unsigned cvt2(float a, float b) {
  union { __bf16 h[2]; unsigned u; } r;
  r.h[0] = (__bf16)a; r.h[1] = (__bf16)b;   // fuses to v_cvt_pk_bf16_f32
  return r.u;
}
__device__ __forceinline__ uint2 cvt4(float4 x) {
  uint2 r; r.x = cvt2(x.x, x.y); r.y = cvt2(x.z, x.w); return r;
}

__device__ __forceinline__ v8f wmma_bf16(v16bf a, v16bf b, v8f c) {
  // D(16x16 f32) = A(16x32 bf16) * B(32x16 bf16) + C
  return __builtin_amdgcn_wmma_f32_16x16x32_bf16(false, a, false, b,
                                                 (short)0, c, false, false);
}

// ---------------------------------------------------------------------------
// Kernel 0: elementwise f32 -> bf16 (for K and V pre-conversion into d_ws)
// ---------------------------------------------------------------------------
__global__ __launch_bounds__(256) void moa_cvt_kernel(
    const float* __restrict__ src, unsigned short* __restrict__ dst) {
  const int i = blockIdx.x * 256 + threadIdx.x;   // one float4 per thread
  ((uint2*)dst)[i] = cvt4(((const float4*)src)[i]);
}

// ---------------------------------------------------------------------------
// Kernel 1: bar_q[b][f] = mean over N of qt[b,:,f]        (grid 8, block 256)
// ---------------------------------------------------------------------------
__global__ __launch_bounds__(256) void moa_barq_kernel(
    const float* __restrict__ qt, float* __restrict__ barq) {
  const int b = blockIdx.x;
  const int t = threadIdx.x;
  const int f = t & 127;
  const int half = t >> 7;
  float s = 0.f;
  const size_t base = ((size_t)b * 2048 + (size_t)half * 1024) * 128 + f;
  for (int r = 0; r < 1024; ++r) s += qt[base + (size_t)r * 128];
  __shared__ float tmp[256];
  tmp[t] = s;
  __syncthreads();
  if (half == 0) barq[b * 128 + f] = (tmp[f] + tmp[f + 128]) * (1.0f / 2048.0f);
}

// ---------------------------------------------------------------------------
// Kernel 2: prior = softmax(kernel @ barq^T, axis=1) flat row-major (16 vals)
// ---------------------------------------------------------------------------
__global__ void moa_prior_kernel(const float* __restrict__ mixk,
                                 const float* __restrict__ barq,
                                 float* __restrict__ prior) {
  __shared__ float lg[16];
  const int t = threadIdx.x;
  if (t < 16) {
    const int mr = t >> 3, bc = t & 7;
    float s = 0.f;
    for (int f = 0; f < 128; ++f) s += mixk[mr * 128 + f] * barq[bc * 128 + f];
    lg[t] = s;
  }
  __syncthreads();
  if (t < 16) {
    const int mr = t >> 3;
    float mx = lg[mr * 8];
    for (int i = 1; i < 8; ++i) mx = fmaxf(mx, lg[mr * 8 + i]);
    float den = 0.f;
    for (int i = 0; i < 8; ++i) den += __expf(lg[mr * 8 + i] - mx);
    prior[t] = __expf(lg[t] - mx) / den;
  }
}

// ---------------------------------------------------------------------------
// Main fused kernel: grid (8 batches, 32 row-blocks of 64), block 256 (8 waves)
// waves 0-3: mixture 0 (features 0..63); waves 4-7: mixture 1 (features 64..127)
// Each wave owns a 16-query-row tile; online softmax over keys in 64-key chunks.
// ---------------------------------------------------------------------------
struct KVBuf { unsigned short k[64 * 128]; unsigned short v[64 * 128]; }; // 32KB

__global__ __launch_bounds__(256) void moa_flash_kernel(
    const float* __restrict__ qt, const unsigned short* __restrict__ ktb,
    const unsigned short* __restrict__ vtb, const float* __restrict__ prior,
    float* __restrict__ out) {
  union Sh {
    KVBuf buf[2];        // 64KB double-buffered bf16 K/V
    float o[64 * 128];   // 32KB epilogue tile (aliases buf[0])
  };
  __shared__ Sh sh;

  const int b    = blockIdx.x;
  const int tid  = threadIdx.x;
  const int lane = tid & 31;
  const int wave = tid >> 5;
  const int m    = wave >> 2;        // mixture
  const int rsub = wave & 3;         // 16-row sub-tile within 64-row block
  const int row_base = blockIdx.y * 64 + rsub * 16;
  const int hl  = (lane < 16) ? 0 : 1;
  const int l16 = lane & 15;

  // per-wave share of a chunk copy: 4 x 512B for K, 4 x 512B for V
  const unsigned coff = (unsigned)((wave * 4) * 32 + lane) * 16u; // byte offset
  const unsigned ldsK[2] = {
      (unsigned)(uintptr_t)&sh.buf[0].k[0], (unsigned)(uintptr_t)&sh.buf[1].k[0]};
  const unsigned ldsV[2] = {
      (unsigned)(uintptr_t)&sh.buf[0].v[0], (unsigned)(uintptr_t)&sh.buf[1].v[0]};

#define ISSUE_ASYNC(bufi, jj)                                                  \
  do {                                                                         \
    const unsigned short* kg = ktb + ((size_t)(b * 2048 + (jj))) * 128;        \
    const unsigned short* vg = vtb + ((size_t)(b * 2048 + (jj))) * 128;        \
    _Pragma("unroll")                                                          \
    for (int i_ = 0; i_ < 4; ++i_) {                                           \
      const unsigned o_ = coff + (unsigned)i_ * 512u;                          \
      asm volatile("global_load_async_to_lds_b128 %0, %1, %2 offset:0"         \
                   :: "v"(ldsK[bufi] + o_), "v"(o_), "s"(kg) : "memory");      \
      asm volatile("global_load_async_to_lds_b128 %0, %1, %2 offset:0"         \
                   :: "v"(ldsV[bufi] + o_), "v"(o_), "s"(vg) : "memory");      \
    }                                                                          \
  } while (0)

  // ---- Q^T B-operands (built once): B[K=feature][N=qrow], lane = qrow, elements = features (table order) ----
  v16bf qb0, qb1;
  {
    const int f0 = m * 64 + hl * 8;               // k-step 0 feature base
    const size_t rb = ((size_t)(b * 2048 + row_base)) * 128;
    BF16x16 u0, u1;
#pragma unroll
    for (int n = 0; n < 16; ++n) {
      const int kk = (n < 8) ? n : 16 + (n - 8);
      u0.s[n] = bf1(qt[rb + (size_t)l16 * 128 + f0 + kk] * QSCALE);
      u1.s[n] = bf1(qt[rb + (size_t)l16 * 128 + f0 + kk + 32] * QSCALE);
    }
    qb0 = u0.v; qb1 = u1.v;
  }

  const v8f vzero = {0.f, 0.f, 0.f, 0.f, 0.f, 0.f, 0.f, 0.f};
  v8f O[8];                        // 16 qrows x 128 features, f32 accumulators
#pragma unroll
  for (int i = 0; i < 8; ++i) O[i] = vzero;
  float run_max = -1e30f, run_sum = 0.f;   // per-lane (qrow = lane&15)

  ISSUE_ASYNC(0, 0);               // kick off first chunk

  for (int j = 0; j < 2048; j += 64) {
    const int cur = (j >> 6) & 1;
    __syncthreads();               // everyone done computing on buf[cur^1]
    if (j + 64 < 2048) {
      ISSUE_ASYNC(cur ^ 1, j + 64);
      asm volatile("s_wait_asynccnt 0x8" ::: "memory"); // cur's 8 copies done
    } else {
      asm volatile("s_wait_asynccnt 0x0" ::: "memory");
    }
    __syncthreads();               // all waves' cur chunk resident in LDS

    const unsigned short* skv = sh.buf[cur].k;
    const unsigned short* svv = sh.buf[cur].v;

    // ---- K A-operands for 4 key-blocks (preload all, then 8 WMMAs) ----
    // lanes 0-15: row=lane, K={0..7,16..23}; lanes 16-31: row=lane-16, K={8..15,24..31}
    v16bf ka[4][2];
    {
      const int fbase = m * 64 + hl * 8;
#pragma unroll
      for (int kb = 0; kb < 4; ++kb) {
        const unsigned short* p = &skv[(kb * 16 + l16) * 128 + fbase];
        BF16x16 u0, u1;
        u0.q[0] = *(const uint4*)(p);      u0.q[1] = *(const uint4*)(p + 16);
        u1.q[0] = *(const uint4*)(p + 32); u1.q[1] = *(const uint4*)(p + 48);
        ka[kb][0] = u0.v; ka[kb][1] = u1.v;
      }
    }
    // ---- S^T = K_tile @ Q^T : D rows = keys, cols = qrows (lanes) ----
    v8f s[4];
#pragma unroll
    for (int kb = 0; kb < 4; ++kb) {
      v8f acc = vzero;
      acc = wmma_bf16(ka[kb][0], qb0, acc);
      acc = wmma_bf16(ka[kb][1], qb1, acc);
      s[kb] = acc;   // lane j: qrow j&15; VGPR g = key kb*16 + g + 8*hl
    }

    // ---- online softmax (per-lane row state) ----
    float cm = -1e30f;
#pragma unroll
    for (int kb = 0; kb < 4; ++kb)
#pragma unroll
      for (int g = 0; g < 8; ++g) cm = fmaxf(cm, s[kb][g]);
    cm = fmaxf(cm, __shfl_xor(cm, 16, 32));
    const float nm = fmaxf(run_max, cm);

    // ---- rescale O only when the running max actually grew (wave-uniform) --
    if (__any((int)(nm > run_max))) {
      const float sc = __expf(run_max - nm);
      run_sum *= sc;
#pragma unroll
      for (int g = 0; g < 8; ++g) {
        const float scg = __shfl(sc, g + hl * 8, 32);
#pragma unroll
        for (int fb = 0; fb < 8; ++fb) O[fb][g] *= scg;
      }
    }
    run_max = nm;

    float e[4][8], ls = 0.f;
#pragma unroll
    for (int kb = 0; kb < 4; ++kb)
#pragma unroll
      for (int g = 0; g < 8; ++g) {
        e[kb][g] = __expf(s[kb][g] - nm);
        ls += e[kb][g];
      }
    ls += __shfl_xor(ls, 16, 32);
    run_sum += ls;

    // ---- exp(P) is already in A-layout (lane j = qrow, keys hl*8+g) ----
    v16bf pA0, pA1;
    {
      BF16x16 u0, u1;
#pragma unroll
      for (int g = 0; g < 8; g += 2) {
        u0.u[g >> 1]       = cvt2(e[0][g], e[0][g + 1]);  // keys  (hl*8)+g
        u0.u[4 + (g >> 1)] = cvt2(e[1][g], e[1][g + 1]);  // keys 16+(hl*8)+g
        u1.u[g >> 1]       = cvt2(e[2][g], e[2][g + 1]);  // keys 32+(hl*8)+g
        u1.u[4 + (g >> 1)] = cvt2(e[3][g], e[3][g + 1]);  // keys 48+(hl*8)+g
      }
      pA0 = u0.v; pA1 = u1.v;
    }
    // ---- O += P @ V : B[K=key][N=feature(lane)], keys in table order ----
#pragma unroll
    for (int fb = 0; fb < 8; ++fb) {
      BF16x16 w0, w1;
#pragma unroll
      for (int s2 = 0; s2 < 16; ++s2) {
        const int kk = (s2 < 8) ? hl * 8 + s2 : 16 + hl * 8 + (s2 - 8);
        w0.s[s2] = svv[kk * 128 + fb * 16 + l16];
        w1.s[s2] = svv[(32 + kk) * 128 + fb * 16 + l16];
      }
      O[fb] = wmma_bf16(pA0, w0.v, O[fb]);
      O[fb] = wmma_bf16(pA1, w1.v, O[fb]);
    }
  }
#undef ISSUE_ASYNC

  // ---- epilogue: scale by prior/rowsum, combine mixtures through LDS ----
  const float pf = prior[b * 2 + m];   // faithful to reference's flat indexing
  const float coef = pf / run_sum;     // per-lane (qrow = lane&15)
  float cg[8];
#pragma unroll
  for (int g = 0; g < 8; ++g) cg[g] = __shfl(coef, g + hl * 8, 32);

  __syncthreads();                      // done with sh.buf, reuse as sh.o
  if (m == 0) {
#pragma unroll
    for (int g = 0; g < 8; ++g) {
      const int rl = rsub * 16 + g + hl * 8;
#pragma unroll
      for (int fb = 0; fb < 8; ++fb)
        sh.o[rl * 128 + fb * 16 + l16] = O[fb][g] * cg[g];
    }
  }
  __syncthreads();
  if (m == 1) {
#pragma unroll
    for (int g = 0; g < 8; ++g) {
      const int rl = rsub * 16 + g + hl * 8;
#pragma unroll
      for (int fb = 0; fb < 8; ++fb)
        sh.o[rl * 128 + fb * 16 + l16] += O[fb][g] * cg[g];
    }
  }
  __syncthreads();
  // ---- stream 64x128 f32 tile to global ----
  {
    float4* go = (float4*)(out + ((size_t)(b * 2048 + blockIdx.y * 64)) * 128);
    const float4* so = (const float4*)sh.o;
#pragma unroll
    for (int i = 0; i < 8; ++i) go[tid + i * 256] = so[tid + i * 256];
  }
}

// ---------------------------------------------------------------------------
// ws layout: [0..1023] barq f32, [1024..1039] prior f32,
//            byte 8192: K bf16 (4MB), byte 8192+4MB: V bf16 (4MB)
// ---------------------------------------------------------------------------
extern "C" void kernel_launch(void* const* d_in, const int* in_sizes, int n_in,
                              void* d_out, int out_size, void* d_ws,
                              size_t ws_size, hipStream_t stream) {
  (void)in_sizes; (void)n_in; (void)out_size; (void)ws_size;
  const float* qt   = (const float*)d_in[0];
  const float* kt   = (const float*)d_in[1];
  const float* vt   = (const float*)d_in[2];
  const float* mixk = (const float*)d_in[3];
  float* out = (float*)d_out;
  float* wsf = (float*)d_ws;
  float* barq  = wsf;                                   // 8*128 floats
  float* prior = wsf + 1024;                            // 16 floats
  unsigned short* kbf = (unsigned short*)(wsf + 2048);  // 2M bf16 (4MB)
  unsigned short* vbf = kbf + 8 * 2048 * 128;           // 2M bf16 (4MB)

  moa_cvt_kernel<<<dim3(2048), dim3(256), 0, stream>>>(kt, kbf);
  moa_cvt_kernel<<<dim3(2048), dim3(256), 0, stream>>>(vt, vbf);
  moa_barq_kernel<<<dim3(8), dim3(256), 0, stream>>>(qt, barq);
  moa_prior_kernel<<<dim3(1), dim3(32), 0, stream>>>(mixk, barq, prior);
  moa_flash_kernel<<<dim3(8, 32), dim3(256), 0, stream>>>(qt, kbf, vbf, prior, out);
}
